// MultiProtoAsConv_59648505806893
// MI455X (gfx1250) — matrix operand order repaired and checked by reading, not compiled, not run on hardware
//
#include <hip/hip_runtime.h>

// ---------------------------------------------------------------------------
// MultiProtoAsConv for MI455X (gfx1250, wave32, WMMA).
// Heavy phase: M=65536 x N=256 x K=256 GEMM in f16 WMMA (f32 accum) with
// B (prototypes) staged in LDS (padded, bank-conflict-free) and a fused
// per-pixel masked softmax + argmax done in-register via shfl_xor.
// ---------------------------------------------------------------------------

typedef __attribute__((ext_vector_type(16))) _Float16 v16h;
typedef __attribute__((ext_vector_type(8)))  _Float16 v8h;
typedef __attribute__((ext_vector_type(8)))  float    v8f;

#define CC     256
#define HW     65536            // 256*256 pixels
#define NEGB   (-1.0e9f)
#define BPAD   264              // LDS row stride in halves: 132 words == 4 mod 64
                                // -> 16 lanes * 4-word b128 reads cover all banks

// ---------------- Phase 1: pooling ----------------------------------------

__global__ __launch_bounds__(256) void k_pool_x(const float* __restrict__ sx,
                                                float* __restrict__ X) {
  int t    = blockIdx.x * 256 + threadIdx.x;   // 0..65535
  int c    = t >> 8;
  int cell = t & 255;
  int py   = cell >> 4, px = cell & 15;
  const float* base = sx + (size_t)c * HW + (size_t)(py * 16) * 256 + px * 16;
  float s = 0.f;
  for (int r = 0; r < 16; ++r) {
    const float4* p = (const float4*)(base + r * 256);
    for (int q = 0; q < 4; ++q) { float4 v = p[q]; s += v.x + v.y + v.z + v.w; }
  }
  X[t] = s * (1.f / 256.f);
}

__global__ __launch_bounds__(256) void k_pool_y(const float* __restrict__ sy,
                                                float* __restrict__ selm) {
  int cell = threadIdx.x;
  int py = cell >> 4, px = cell & 15;
  const float* base = sy + (size_t)(py * 16) * 256 + px * 16;
  float s = 0.f;
  for (int r = 0; r < 16; ++r) {
    const float4* p = (const float4*)(base + r * 256);
    for (int q = 0; q < 4; ++q) { float4 v = p[q]; s += v.x + v.y + v.z + v.w; }
  }
  selm[cell] = (s * (1.f / 256.f) > 0.5f) ? 1.f : 0.f;
}

// ---------------- Phase 2: prototype construction (cache-resident) --------

__global__ __launch_bounds__(256) void k_gram(const float* __restrict__ X,
                                              float* __restrict__ w1) {
  int t = blockIdx.x * 256 + threadIdx.x;
  int i = t >> 8, j = t & 255;
  const float* ri = X + i * 256;
  const float* rj = X + j * 256;
  float s = 0.f;
  for (int p = 0; p < 256; ++p) s += ri[p] * rj[p];
  w1[t] = s;
}

__global__ __launch_bounds__(256) void k_soft_A(const float* __restrict__ w1,
                                                const float* __restrict__ cal,
                                                float* __restrict__ M2) {
  __shared__ float red[256];
  int i = blockIdx.x, j = threadIdx.x;
  float v = w1[i * 256 + j];
  red[j] = v; __syncthreads();
  for (int s = 128; s > 0; s >>= 1) {
    if (j < s) red[j] = fmaxf(red[j], red[j + s]);
    __syncthreads();
  }
  float mx = red[0]; __syncthreads();
  float e = __expf(v - mx);
  red[j] = e; __syncthreads();
  for (int s = 128; s > 0; s >>= 1) {
    if (j < s) red[j] += red[j + s];
    __syncthreads();
  }
  float soft = e / red[0];
  int d = i - j;
  float maskw = (d <= 1 && d >= -1) ? 1.f : 0.f;
  M2[i * 256 + j] = (1.f + 0.2f * maskw * soft) * cal[i * 256 + j];
}

__global__ __launch_bounds__(256) void k_xn(const float* __restrict__ M2,
                                            const float* __restrict__ X,
                                            float* __restrict__ Xn) {
  int t = blockIdx.x * 256 + threadIdx.x;
  int i = t >> 8, pos = t & 255;
  const float* row = M2 + i * 256;
  float s = 0.f;
  for (int j = 0; j < 256; ++j) s += row[j] * X[j * 256 + pos];
  Xn[t] = s;
}

// proto16[n][c] = f16( Xn[c][n] / max(||Xn[:,n]||, 1e-4) )
__global__ __launch_bounds__(256) void k_proto(const float* __restrict__ Xn,
                                               _Float16* __restrict__ p16) {
  __shared__ float red[256];
  int n = blockIdx.x, c = threadIdx.x;
  float v = Xn[c * 256 + n];
  red[c] = v * v; __syncthreads();
  for (int s = 128; s > 0; s >>= 1) {
    if (c < s) red[c] += red[c + s];
    __syncthreads();
  }
  float rn = 1.f / fmaxf(sqrtf(red[0]), 1e-4f);
  p16[n * 256 + c] = (_Float16)(v * rn);
}

// ---------------- Phase 3a: per-pixel inverse norms of qry ----------------

__global__ __launch_bounds__(256) void k_rnorm(const float* __restrict__ q,
                                               float* __restrict__ rn) {
  int p = blockIdx.x * 256 + threadIdx.x;     // coalesced across p per c
  float s = 0.f;
  for (int c = 0; c < 256; ++c) {
    float v = q[(size_t)c * HW + p];
    s += v * v;
  }
  rn[p] = 1.f / fmaxf(sqrtf(s), 1e-4f);
}

// ---------------- Phase 3b: transpose + normalize + f32->f16 --------------
// channel-major f32 -> pixel-major f16 via LDS tile (64 pixels x 256 ch).

__global__ __launch_bounds__(256) void k_qn16(const float* __restrict__ q,
                                              const float* __restrict__ rn,
                                              unsigned* __restrict__ out) {
  __shared__ float lds[64 * 257];
  int t  = threadIdx.x;
  int p0 = blockIdx.x * 64;
  for (int j = 0; j < 64; ++j) {
    int idx = j * 256 + t;
    int c   = idx >> 6;
    int pp  = idx & 63;
    lds[pp * 257 + c] = q[(size_t)c * HW + p0 + pp] * rn[p0 + pp];
  }
  __syncthreads();
  for (int j2 = 0; j2 < 32; ++j2) {
    int o   = j2 * 256 + t;
    int row = o >> 7;              // 0..63
    int ci  = (o & 127) * 2;       // 0..254 even
    union { _Float16 h[2]; unsigned u; } pk;
    pk.h[0] = (_Float16)lds[row * 257 + ci];
    pk.h[1] = (_Float16)lds[row * 257 + ci + 1];
    out[(size_t)(p0 + row) * 128 + (o & 127)] = pk.u;   // qn16 pixel-major
  }
}

// ---------------- Phase 4: WMMA GEMM + fused masked softmax/argmax --------
// Each wave: 16 pixels x 256 protos, K=256 in 8 steps of 16x16x32 f16 WMMA.
// B (128 KB of prototypes) is staged once per workgroup into LDS with a
// padded row stride so the per-tile ds_load_b128s are bank-conflict-free.
// C/D layout: lane L holds col N = ntile*16 + (L&15), rows m0+(L>>4)*8 + j.

__global__ __launch_bounds__(256) void k_gemm_softmax(
    const _Float16* __restrict__ qn16,   // [HW][256] pixel-major, normalized
    const _Float16* __restrict__ p16,    // [256 protos][256 ch]
    const float*    __restrict__ selm,   // [256] 0/1
    float* __restrict__ out) {           // [0,HW): pred, [HW,2HW): argmax
  __shared__ _Float16 Blds[256 * BPAD];  // 132 KB

  const int tid  = threadIdx.x;
  const int lane = tid & 31;
  const int wave = tid >> 5;
  const int m0   = (blockIdx.x * 8 + wave) * 16;
  const int lrow = lane & 15;
  const int lhi  = lane >> 4;

  // --- Stage B: 256 rows x 256 halves, coalesced 16B chunks per lane. ---
#pragma unroll 4
  for (int i = 0; i < 32; ++i) {
    int chunk = i * 256 + tid;          // 0..8191
    int row   = chunk >> 5;             // 0..255
    int c8    = (chunk & 31) * 8;       // half offset 0..248
    v8h v = *(const v8h*)(p16 + (size_t)row * 256 + c8);
    *(v8h*)(Blds + row * BPAD + c8) = v;
  }
  __syncthreads();

  v8f acc[16];
#pragma unroll
  for (int n = 0; n < 16; ++n)
#pragma unroll
    for (int j = 0; j < 8; ++j) acc[n][j] = 0.f;

  const _Float16* qrow = qn16 + (size_t)(m0 + lrow) * 256;

  for (int k0 = 0; k0 < 256; k0 += 32) {
    // Prefetch next k-step of the A stream (global_prefetch_b8 path).
    if (k0 + 32 < 256) __builtin_prefetch(qrow + k0 + 32, 0, 0);

    // A tile 16x32 f16: interleaved per-lane layout (ISA 7.12.2):
    // lane holds row (lane&15); halves 0..7 = K[k0+lhi*8..], 8..15 = K[k0+16+lhi*8..]
    v8h alo = *(const v8h*)(qrow + k0 + lhi * 8);
    v8h ahi = *(const v8h*)(qrow + k0 + 16 + lhi * 8);
    v16h a;
#pragma unroll
    for (int i = 0; i < 8; ++i) { a[i] = alo[i]; a[8 + i] = ahi[i]; }

#pragma unroll
    for (int n = 0; n < 16; ++n) {
      // B tile 32x16 f16 from LDS: lane holds col (lane&15); same interleave as A:
      // halves 0..7 = K[k0+lhi*8..], 8..15 = K[k0+16+lhi*8..].
      const _Float16* bp = Blds + (n * 16 + lrow) * BPAD + k0 + lhi * 8;
      v8h blo = *(const v8h*)bp;
      v8h bhi = *(const v8h*)(bp + 16);
      v16h b;
#pragma unroll
      for (int i = 0; i < 8; ++i) { b[i] = blo[i]; b[8 + i] = bhi[i]; }
      acc[n] = __builtin_amdgcn_wmma_f32_16x16x32_f16(
          false, a, false, b, (short)0, acc[n], false, false);
    }
  }

  // Mask + scale (dists = 20 * cos, NEGB where !sel).
#pragma unroll
  for (int n = 0; n < 16; ++n) {
    float s = selm[n * 16 + lrow];
#pragma unroll
    for (int j = 0; j < 8; ++j) {
      float l = acc[n][j] * 20.f;
      acc[n][j] = (s > 0.5f) ? l : NEGB;
    }
  }

  // Row-wise max + argmax (first-max tie-break like jnp.argmax).
  float mx[8]; int ax[8];
#pragma unroll
  for (int j = 0; j < 8; ++j) { mx[j] = -3.4e38f; ax[j] = 0x7fffffff; }
#pragma unroll
  for (int n = 0; n < 16; ++n) {
    int col = n * 16 + lrow;
#pragma unroll
    for (int j = 0; j < 8; ++j) {
      float l = acc[n][j];
      if (l > mx[j] || (l == mx[j] && col < ax[j])) { mx[j] = l; ax[j] = col; }
    }
  }
#pragma unroll
  for (int off = 1; off <= 8; off <<= 1) {
#pragma unroll
    for (int j = 0; j < 8; ++j) {
      float ov = __shfl_xor(mx[j], off, 32);
      int   oi = __shfl_xor(ax[j], off, 32);
      if (ov > mx[j] || (ov == mx[j] && oi < ax[j])) { mx[j] = ov; ax[j] = oi; }
    }
  }

  // Softmax sums: se = sum(exp(l-mx)), sl = sum(exp(l-mx)*l).
  float se[8], sl[8];
#pragma unroll
  for (int j = 0; j < 8; ++j) { se[j] = 0.f; sl[j] = 0.f; }
#pragma unroll
  for (int n = 0; n < 16; ++n)
#pragma unroll
    for (int j = 0; j < 8; ++j) {
      float l = acc[n][j];
      float e = __expf(l - mx[j]);
      se[j] += e; sl[j] += e * l;
    }
#pragma unroll
  for (int off = 1; off <= 8; off <<= 1) {
#pragma unroll
    for (int j = 0; j < 8; ++j) {
      se[j] += __shfl_xor(se[j], off, 32);
      sl[j] += __shfl_xor(sl[j], off, 32);
    }
  }

  if (lrow == 0) {
    int rbase = m0 + lhi * 8;
#pragma unroll
    for (int j = 0; j < 8; ++j) {
      out[rbase + j]      = sl[j] / se[j];
      out[HW + rbase + j] = (float)ax[j];
    }
  }
}

// ---------------------------------------------------------------------------

extern "C" void kernel_launch(void* const* d_in, const int* in_sizes, int n_in,
                              void* d_out, int out_size, void* d_ws, size_t ws_size,
                              hipStream_t stream) {
  (void)in_sizes; (void)n_in; (void)out_size; (void)ws_size;
  const float* qry   = (const float*)d_in[0];  // (C, 256, 256)
  const float* sup_x = (const float*)d_in[1];  // (C, 256, 256)
  const float* sup_y = (const float*)d_in[2];  // (256, 256)
  /* d_in[3] s_init_seed unused by the reference */
  const float* cal   = (const float*)d_in[4];  // (256, 256)

  char* ws = (char*)d_ws;
  float*    X      = (float*)ws;            ws += (size_t)CC * 256 * 4;   // 256KB
  float*    selm   = (float*)ws;            ws += 1024;                   // 1KB
  float*    w1     = (float*)ws;            ws += (size_t)CC * 256 * 4;   // 256KB
  float*    M2     = (float*)ws;            ws += (size_t)CC * 256 * 4;   // 256KB
  float*    Xn     = (float*)ws;            ws += (size_t)CC * 256 * 4;   // 256KB
  _Float16* p16    = (_Float16*)ws;         ws += (size_t)CC * 256 * 2;   // 128KB
  float*    rnorm  = (float*)ws;            ws += (size_t)HW * 4;         // 256KB
  _Float16* qn16   = (_Float16*)ws;         ws += (size_t)HW * 256 * 2;   // 32MB

  k_pool_x<<<256, 256, 0, stream>>>(sup_x, X);
  k_pool_y<<<1,   256, 0, stream>>>(sup_y, selm);
  k_gram  <<<256, 256, 0, stream>>>(X, w1);
  k_soft_A<<<256, 256, 0, stream>>>(w1, cal, M2);
  k_xn    <<<256, 256, 0, stream>>>(M2, X, Xn);
  k_proto <<<256, 256, 0, stream>>>(Xn, p16);
  k_rnorm <<<256, 256, 0, stream>>>(qry, rnorm);
  k_qn16  <<<1024,256, 0, stream>>>(qry, rnorm, (unsigned*)qn16);
  k_gemm_softmax<<<512, 256, 0, stream>>>(qn16, p16, selm, (float*)d_out);
}
